// Cylinder3D_6408091206414
// MI455X (gfx1250) — matrix operand order repaired and checked by reading, not compiled, not run on hardware
//
#include <hip/hip_runtime.h>
#include <hip/hip_bf16.h>

typedef __attribute__((ext_vector_type(16))) __bf16 v16bf;
typedef __attribute__((ext_vector_type(8)))  float  v8f;

#define COUT 64
#define KOFF 9

union PkBf2 { __bf16 h[2]; unsigned u; };
union FragA { v16bf v; uint4 q[2]; };
union FragB { v16bf v; uint4 q[2]; };

// ---------------------------------------------------------------------------
// Fused gather(bf16) -> bf16 WMMA GEMM -> LeakyReLU -> store + stats.
// Inputs are pre-normalized bf16 with a zero row appended at index n; masked
// neighbors are redirected to the zero row at index-preload time, so the
// A-fragment build is just two 16-B loads per 4 WMMAs (no masking VALU).
// Weights are staged in LDS pre-swizzled into per-lane B-fragment order
// (32B contiguous per lane). Per-channel BN statistics are reduced through
// LDS before touching global atomics.
// ---------------------------------------------------------------------------
template<int CIN>
__global__ __launch_bounds__(256)
void conv_bn_kernel(const unsigned short* __restrict__ Xb, // [n+1, CIN] bf16
                    const int*   __restrict__ nbr,         // [KOFF, n]
                    const unsigned char* __restrict__ mask,// [KOFF, n] bool
                    const float* __restrict__ W,           // [KOFF, CIN, COUT]
                    float*       __restrict__ Y,           // [n, COUT] lrelu out
                    float*       __restrict__ stats,       // [2*COUT] sum,sumsq
                    int n)
{
    constexpr int CH = CIN / 32;                 // 32-wide K chunks
    constexpr int KS = (CIN == 32) ? 9 : 5;      // k-offsets resident in LDS
    constexpr int WFR_UINTS = KS * CH * 128 * 8; // weight fragment uints
    extern __shared__ unsigned wlds[];           // fragments: 8 uints (32B) each
    float* sstat = (float*)(wlds + WFR_UINTS);   // [128] block-local stats

    const int tid  = threadIdx.x;
    const int lane = tid & 31;
    const int wave = tid >> 5;
    const int half = lane >> 4;                  // 0: lanes 0-15, 1: lanes 16-31
    const int mrow = lane & 15;                  // row within 16-row tile
    const int m0   = blockIdx.x * 128 + wave * 16;
    const int m    = m0 + mrow;
    const int mc   = (m < n) ? m : (n - 1);      // clamp so EXEC stays full

    if (tid < 128) sstat[tid] = 0.f;             // covered by first barrier

    // Preload neighbor indices; masked-off neighbors point at the zero row n
    int idxs[KOFF];
#pragma unroll
    for (int k = 0; k < KOFF; ++k) {
        const int ix = nbr[(size_t)k * n + mc];
        idxs[k] = mask[(size_t)k * n + mc] ? ix : n;
    }

    v8f acc[4];
#pragma unroll
    for (int td = 0; td < 4; ++td)
        acc[td] = (v8f){0.f, 0.f, 0.f, 0.f, 0.f, 0.f, 0.f, 0.f};

#pragma unroll
    for (int k0 = 0; k0 < KOFF; k0 += KS) {
        const int kn = (KOFF - k0 < KS) ? (KOFF - k0) : KS;

        __syncthreads();   // previous LDS pass fully consumed
        // -------- stage weights for k in [k0, k0+kn), bf16, fragment order
        for (int f = tid; f < kn * CH * 128; f += 256) {
            int fl = f & 31;
            int t  = f >> 5;
            int td = t & 3;  t >>= 2;
            int ch = t % CH;
            int k  = t / CH;
            const int crow = ch * 32 + ((fl >> 4) << 4);      // 16*(lane/16)
            const int col  = td * 16 + (fl & 15);
            const float* wp = W + ((size_t)(k0 + k) * CIN + crow) * COUT + col;
#pragma unroll
            for (int j2 = 0; j2 < 8; ++j2) {
                PkBf2 pk;
                pk.h[0] = (__bf16)wp[(2 * j2) * COUT];
                pk.h[1] = (__bf16)wp[(2 * j2 + 1) * COUT];
                wlds[f * 8 + j2] = pk.u;
            }
        }
        __syncthreads();

        // -------- accumulate this k-slab
#pragma unroll
        for (int kl = 0; kl < kn; ++kl) {
            const int kk = k0 + kl;
            // prefetch next k's gather row into cache (global_prefetch_b8)
            if (kl + 1 < kn)
                __builtin_prefetch(Xb + (size_t)idxs[kk + 1] * CIN + half * 8, 0, 1);
            const unsigned short* src = Xb + (size_t)idxs[kk] * CIN;
#pragma unroll
            for (int ch = 0; ch < CH; ++ch) {
                const int cbase = ch * 32 + half * 16;
                const uint4* p = reinterpret_cast<const uint4*>(src + cbase);
                FragA A;
                A.q[0] = p[0];   // bf16 elems cbase    .. cbase+7
                A.q[1] = p[1];   // bf16 elems cbase+8  .. cbase+15
#pragma unroll
                for (int td = 0; td < 4; ++td) {
                    FragB B;
                    const uint4* bp = reinterpret_cast<const uint4*>(
                        &wlds[(((size_t)((kl * CH + ch) * 4 + td) * 32 + lane) * 8)]);
                    B.q[0] = bp[0];
                    B.q[1] = bp[1];
                    acc[td] = __builtin_amdgcn_wmma_f32_16x16x32_bf16(
                        false, A.v, false, B.v, (short)0, acc[td], false, false);
                }
            }
        }
    }

    // -------- epilogue: LeakyReLU in-place
#pragma unroll
    for (int td = 0; td < 4; ++td)
#pragma unroll
        for (int r = 0; r < 8; ++r) {
            float x = acc[td][r];
            acc[td][r] = (x > 0.f) ? x : 0.01f * x;
        }

    // store + per-channel sum / sumsq (LDS reduction first)
    if (m0 + 16 <= n) {
        // fast path: full tile, no per-row guards
#pragma unroll
        for (int td = 0; td < 4; ++td) {
            const int d = td * 16 + mrow;
            float s1 = 0.f, s2 = 0.f;
#pragma unroll
            for (int r = 0; r < 8; ++r) {
                const float x = acc[td][r];
                const int mm = m0 + r + half * 8;      // C-matrix row layout
                Y[(size_t)mm * COUT + d] = x;
                s1 += x;
                s2 += x * x;
            }
            s1 += __shfl_xor(s1, 16, 32);
            s2 += __shfl_xor(s2, 16, 32);
            if (half == 0) {
                atomicAdd(&sstat[d], s1);
                atomicAdd(&sstat[COUT + d], s2);
            }
        }
    } else {
        // slow path: only ever the final partial block
#pragma unroll
        for (int td = 0; td < 4; ++td) {
            const int d = td * 16 + mrow;
            float s1 = 0.f, s2 = 0.f;
#pragma unroll
            for (int r = 0; r < 8; ++r) {
                const float x = acc[td][r];
                const int mm = m0 + r + half * 8;
                if (mm < n) {
                    Y[(size_t)mm * COUT + d] = x;
                    s1 += x;
                    s2 += x * x;
                }
            }
            s1 += __shfl_xor(s1, 16, 32);
            s2 += __shfl_xor(s2, 16, 32);
            if (half == 0) {
                atomicAdd(&sstat[d], s1);
                atomicAdd(&sstat[COUT + d], s2);
            }
        }
    }

    __syncthreads();
    if (tid < 128) atomicAdd(&stats[tid], sstat[tid]);   // 128 global atomics/block
}

// ---------------------------------------------------------------------------
// zero stats + write the zero rows of both bf16 buffers (row n)
__global__ void init_ws_kernel(float* __restrict__ stats_all,      // 512 floats
                               unsigned short* __restrict__ xzrow, // 32 bf16
                               unsigned short* __restrict__ zzrow) // 64 bf16
{
    const int t = threadIdx.x;
    if (t < 512) stats_all[t] = 0.f;
    if (t < 32) xzrow[t] = 0;
    if (t < 64) zzrow[t] = 0;
}

// f32 -> bf16 streaming convert (8 elems/thread)
__global__ void cvt_bf16_kernel(const float* __restrict__ in,
                                unsigned short* __restrict__ out, long total)
{
    const long i0 = ((long)blockIdx.x * blockDim.x + threadIdx.x) * 8;
    const long stride = (long)gridDim.x * blockDim.x * 8;
    for (long i = i0; i < total; i += stride) {
        const float4 a = *reinterpret_cast<const float4*>(in + i);
        const float4 b = *reinterpret_cast<const float4*>(in + i + 4);
        PkBf2 p0, p1, p2, p3;
        p0.h[0] = (__bf16)a.x; p0.h[1] = (__bf16)a.y;
        p1.h[0] = (__bf16)a.z; p1.h[1] = (__bf16)a.w;
        p2.h[0] = (__bf16)b.x; p2.h[1] = (__bf16)b.y;
        p3.h[0] = (__bf16)b.z; p3.h[1] = (__bf16)b.w;
        uint4 o = {p0.u, p1.u, p2.u, p3.u};
        *reinterpret_cast<uint4*>(out + i) = o;
    }
}

// y = bn(x) as bf16 (8 elems/thread, 8 | COUT so channels stay in-row)
__global__ void bn_apply_cvt_kernel(const float* __restrict__ in,   // [n, 64]
                                    const float* __restrict__ scale,
                                    const float* __restrict__ shift,
                                    unsigned short* __restrict__ out,
                                    long total)
{
    const long i0 = ((long)blockIdx.x * blockDim.x + threadIdx.x) * 8;
    const long stride = (long)gridDim.x * blockDim.x * 8;
    for (long i = i0; i < total; i += stride) {
        const int d0 = (int)(i & 63);
        float v[8];
#pragma unroll
        for (int j = 0; j < 8; ++j)
            v[j] = in[i + j] * scale[d0 + j] + shift[d0 + j];
        PkBf2 p0, p1, p2, p3;
        p0.h[0] = (__bf16)v[0]; p0.h[1] = (__bf16)v[1];
        p1.h[0] = (__bf16)v[2]; p1.h[1] = (__bf16)v[3];
        p2.h[0] = (__bf16)v[4]; p2.h[1] = (__bf16)v[5];
        p3.h[0] = (__bf16)v[6]; p3.h[1] = (__bf16)v[7];
        uint4 o = {p0.u, p1.u, p2.u, p3.u};
        *reinterpret_cast<uint4*>(out + i) = o;
    }
}

__global__ void bn_finalize_kernel(const float* __restrict__ stats,
                                   const float* __restrict__ gamma,
                                   const float* __restrict__ beta,
                                   float* __restrict__ scale,
                                   float* __restrict__ shift,
                                   float inv_n)
{
    const int d  = threadIdx.x;               // 64 threads
    const float mu  = stats[d] * inv_n;
    const float var = stats[COUT + d] * inv_n - mu * mu;
    const float rs  = rsqrtf(var + 1e-5f);
    const float sc  = gamma[d] * rs;
    scale[d] = sc;
    shift[d] = beta[d] - mu * sc;
}

__global__ void combine_kernel(float* __restrict__ out,          // r pre-BN (in place)
                               const float* __restrict__ sbuf,   // s pre-BN
                               const float* __restrict__ scr, const float* __restrict__ shr,
                               const float* __restrict__ scs, const float* __restrict__ shs,
                               int n)
{
    const long total  = (long)n * COUT;
    const long stride = (long)gridDim.x * blockDim.x;
    for (long i = (long)blockIdx.x * blockDim.x + threadIdx.x; i < total; i += stride) {
        const int d = (int)(i & 63);
        out[i] = out[i] * scr[d] + shr[d] + sbuf[i] * scs[d] + shs[d];
    }
}

// ---------------------------------------------------------------------------
extern "C" void kernel_launch(void* const* d_in, const int* in_sizes, int n_in,
                              void* d_out, int out_size, void* d_ws, size_t ws_size,
                              hipStream_t stream)
{
    const float* feats = (const float*)d_in[0];
    const float* W1    = (const float*)d_in[1];
    const float* W12   = (const float*)d_in[2];
    const float* W2    = (const float*)d_in[3];
    const float* W3    = (const float*)d_in[4];
    const float* g0  = (const float*)d_in[5];
    const float* b0  = (const float*)d_in[6];
    const float* g02 = (const float*)d_in[7];
    const float* b02 = (const float*)d_in[8];
    const float* g1  = (const float*)d_in[9];
    const float* b1  = (const float*)d_in[10];
    const float* g2  = (const float*)d_in[11];
    const float* b2  = (const float*)d_in[12];
    const int*  nbrA = (const int*)d_in[13];
    const unsigned char* maskA = (const unsigned char*)d_in[14];
    const int*  nbrB = (const int*)d_in[15];
    const unsigned char* maskB = (const unsigned char*)d_in[16];

    const int n = in_sizes[0] / 32;

    float* wsf     = (float*)d_ws;
    float* stats0  = wsf + 0;
    float* stats02 = wsf + 128;
    float* stats1  = wsf + 256;
    float* stats2  = wsf + 384;
    float* sc0  = wsf + 512,  *sh0  = wsf + 576;
    float* sc02 = wsf + 640,  *sh02 = wsf + 704;
    float* sc1  = wsf + 768,  *sh1  = wsf + 832;
    float* sc2  = wsf + 896,  *sh2  = wsf + 960;
    float* bufF  = wsf + 1024;              // S1/R1 pre-BN f32      [n*64]
    float* bufS2 = bufF + (size_t)n * COUT; // S2 pre-BN f32         [n*64]
    unsigned short* xbf = (unsigned short*)(bufS2 + (size_t)n * COUT); // [(n+1)*32]
    unsigned short* zbf = xbf + (size_t)(n + 1) * 32;                  // [(n+1)*64]
    float* r2pre = (float*)d_out;           // R2 pre-BN staged in d_out

    const int nb = (n + 127) / 128;
    const size_t sh32 = 9 * 1 * 128 * 32 + 512;   // 36 KB weights + 512B stats
    const size_t sh64 = 5 * 2 * 128 * 32 + 512;   // 40 KB weights + 512B stats
    const float inv_n = 1.0f / (float)n;
    const long  tot32 = (long)n * 32;
    const long  tot64 = (long)n * COUT;

    init_ws_kernel<<<1, 512, 0, stream>>>(wsf, xbf + tot32, zbf + tot64);
    cvt_bf16_kernel<<<1024, 256, 0, stream>>>(feats, xbf, tot32);

    // shortcut branch: convA(W1) -> bn0 -> convB(W12) -> (bn02 at combine)
    conv_bn_kernel<32><<<nb, 256, sh32, stream>>>(xbf, nbrA, maskA, W1,
                                                  bufF, stats0, n);
    bn_finalize_kernel<<<1, 64, 0, stream>>>(stats0, g0, b0, sc0, sh0, inv_n);
    bn_apply_cvt_kernel<<<1024, 256, 0, stream>>>(bufF, sc0, sh0, zbf, tot64);
    conv_bn_kernel<64><<<nb, 256, sh64, stream>>>(zbf, nbrB, maskB, W12,
                                                  bufS2, stats02, n);
    bn_finalize_kernel<<<1, 64, 0, stream>>>(stats02, g02, b02, sc02, sh02, inv_n);

    // resA branch: convB(W2) -> bn1 -> convA(W3) -> (bn2 at combine)
    conv_bn_kernel<32><<<nb, 256, sh32, stream>>>(xbf, nbrB, maskB, W2,
                                                  bufF, stats1, n);
    bn_finalize_kernel<<<1, 64, 0, stream>>>(stats1, g1, b1, sc1, sh1, inv_n);
    bn_apply_cvt_kernel<<<1024, 256, 0, stream>>>(bufF, sc1, sh1, zbf, tot64);
    conv_bn_kernel<64><<<nb, 256, sh64, stream>>>(zbf, nbrA, maskA, W3,
                                                  r2pre, stats2, n);
    bn_finalize_kernel<<<1, 64, 0, stream>>>(stats2, g2, b2, sc2, sh2, inv_n);

    // out = bn2(r2pre) + bn02(s2)
    combine_kernel<<<2048, 256, 0, stream>>>(r2pre, bufS2, sc2, sh2, sc02, sh02, n);
}
